// GraphModelCustom_26525718020380
// MI455X (gfx1250) — matrix operand order repaired and checked by reading, not compiled, not run on hardware
//
#include <hip/hip_runtime.h>
#include <hip/hip_bf16.h>

#define NCONN 50000
#define NEDGE 400000
#define EPSV  1e-5f

typedef __attribute__((ext_vector_type(16))) _Float16 v16h;
typedef __attribute__((ext_vector_type(8)))  _Float16 v8h;
typedef __attribute__((ext_vector_type(4)))  _Float16 v4h;
typedef __attribute__((ext_vector_type(8)))  float    v8f;

union U16 { v16h v; v8h h[2]; };

static inline int cdiv(long long a, long long b) { return (int)((a + b - 1) / b); }

// ---------------------------------------------------------------------------
// Generic direct convolution (NCHW, OIHW weights), f32.
// ---------------------------------------------------------------------------
__global__ void k_conv(const float* __restrict__ in, const float* __restrict__ w,
                       float* __restrict__ out,
                       int Cin, int Hin, int Win, int Cout, int Hout, int Wout,
                       int ks, int stride, int pad) {
  size_t idx = blockIdx.x * (size_t)blockDim.x + threadIdx.x;
  size_t total = (size_t)Cout * Hout * Wout;
  if (idx >= total) return;
  int ow = (int)(idx % Wout);
  int oh = (int)((idx / Wout) % Hout);
  int co = (int)(idx / ((size_t)Wout * Hout));
  float acc = 0.f;
  for (int ci = 0; ci < Cin; ++ci) {
    const float* inc = in + (size_t)ci * Hin * Win;
    const float* wc  = w + (((size_t)co * Cin + ci) * ks) * ks;
    for (int kh = 0; kh < ks; ++kh) {
      int ih = oh * stride - pad + kh;
      if (ih < 0 || ih >= Hin) continue;
      for (int kw = 0; kw < ks; ++kw) {
        int iw = ow * stride - pad + kw;
        if (iw < 0 || iw >= Win) continue;
        acc += inc[(size_t)ih * Win + iw] * wc[kh * ks + kw];
      }
    }
  }
  out[idx] = acc;
}

// ---------------------------------------------------------------------------
// Per-channel mean / inv-std over spatial (train-mode BN2d stats).
// ---------------------------------------------------------------------------
__global__ void k_chan_stats(const float* __restrict__ in, float* __restrict__ stats, int HW) {
  int c = blockIdx.x;
  const float* p = in + (size_t)c * HW;
  float s = 0.f, s2 = 0.f;
  for (int i = threadIdx.x; i < HW; i += blockDim.x) { float v = p[i]; s += v; s2 += v * v; }
  __shared__ float shs[256];
  __shared__ float shq[256];
  int tid = threadIdx.x;
  shs[tid] = s; shq[tid] = s2;
  __syncthreads();
  for (int off = 128; off > 0; off >>= 1) {
    if (tid < off) { shs[tid] += shs[tid + off]; shq[tid] += shq[tid + off]; }
    __syncthreads();
  }
  if (tid == 0) {
    float m = shs[0] / (float)HW;
    float v = shq[0] / (float)HW - m * m;
    stats[2 * c]     = m;
    stats[2 * c + 1] = rsqrtf(v + EPSV);
  }
}

// BN2d apply + optional residual add + optional ReLU.
__global__ void k_bn_act(const float* __restrict__ in, const float* __restrict__ stats,
                         const float* __restrict__ g, const float* __restrict__ b,
                         const float* __restrict__ res, float* __restrict__ out,
                         int C, int HW, int relu) {
  size_t idx = blockIdx.x * (size_t)blockDim.x + threadIdx.x;
  size_t total = (size_t)C * HW;
  if (idx >= total) return;
  int c = (int)(idx / HW);
  float y = (in[idx] - stats[2 * c]) * stats[2 * c + 1] * g[c] + b[c];
  if (res) y += res[idx];
  if (relu) y = fmaxf(y, 0.f);
  out[idx] = y;
}

// 3x3 s2 p1 max pool.
__global__ void k_maxpool(const float* __restrict__ in, float* __restrict__ out,
                          int C, int Hin, int Win, int Hout, int Wout) {
  size_t idx = blockIdx.x * (size_t)blockDim.x + threadIdx.x;
  size_t total = (size_t)C * Hout * Wout;
  if (idx >= total) return;
  int ow = (int)(idx % Wout);
  int oh = (int)((idx / Wout) % Hout);
  int c  = (int)(idx / ((size_t)Wout * Hout));
  float m = -INFINITY;
  for (int kh = 0; kh < 3; ++kh) {
    int ih = oh * 2 - 1 + kh;
    if (ih < 0 || ih >= Hin) continue;
    for (int kw = 0; kw < 3; ++kw) {
      int iw = ow * 2 - 1 + kw;
      if (iw < 0 || iw >= Win) continue;
      m = fmaxf(m, in[((size_t)c * Hin + ih) * Win + iw]);
    }
  }
  out[idx] = m;
}

__global__ void k_copy(const float* __restrict__ in, float* __restrict__ out, size_t n) {
  size_t idx = blockIdx.x * (size_t)blockDim.x + threadIdx.x;
  if (idx < n) out[idx] = in[idx];
}

// Vectorized f32 -> f16 (n must be a multiple of 4): b128 load, b64 store.
__global__ void k_f32_to_f16v4(const float* __restrict__ in, _Float16* __restrict__ out, size_t n4) {
  size_t i = blockIdx.x * (size_t)blockDim.x + threadIdx.x;
  if (i >= n4) return;
  float4 v = ((const float4*)in)[i];
  v4h h = { (_Float16)v.x, (_Float16)v.y, (_Float16)v.z, (_Float16)v.w };
  ((v4h*)out)[i] = h;
}

// ---------------------------------------------------------------------------
// Pack B[K,O] (f32 row-major) into WMMA fragment order (f16):
//   Bp[(kt*O + col)*32 + kk] = B[kt*32 + kk, col]
// so each column's 32 K are contiguous; lane (col,hf) reads K = hf*8+[0,8) and 16+hf*8+[0,8).
// ---------------------------------------------------------------------------
__global__ void k_pack_b(const float* __restrict__ w, _Float16* __restrict__ bp, int K, int O) {
  size_t idx = blockIdx.x * (size_t)blockDim.x + threadIdx.x;
  size_t total = (size_t)K * O;
  if (idx >= total) return;
  int kk = (int)(idx & 31);
  size_t rest = idx >> 5;
  int col = (int)(rest % O);
  int kt  = (int)(rest / O);
  bp[idx] = (_Float16)w[(size_t)(kt * 32 + kk) * O + col];
}

// ---------------------------------------------------------------------------
// WMMA strip GEMM: C[M,O] = A[M,K] @ B[K,O]. One wave32 owns a 16-row strip
// and all TN = O/16 column tiles (accumulators stay in VGPRs -> A is read
// exactly once per layer). A fragment = two aligned b128 loads per K-step
// (ISA 7.12.2: lane(r,hf) needs K = hf*8+[0,8) and 16+hf*8+[0,8)).
// B fragment = two b128 loads from the packed layout (same K order as A).
// M = 50000 = 3125*16, all K multiples of 32 -> no edge guards.
// ---------------------------------------------------------------------------
template <int TN>
__global__ void k_wmma_strip(const _Float16* __restrict__ A, const _Float16* __restrict__ Bp,
                             float* __restrict__ C, int M, int K, int O) {
  int wave = (int)((blockIdx.x * (size_t)blockDim.x + threadIdx.x) >> 5);
  int lane = threadIdx.x & 31;
  if (wave >= (M >> 4)) return;
  int r  = lane & 15;
  int hf = lane >> 4;

  const _Float16* arow = A + (size_t)(wave * 16 + r) * K + hf * 8;
  v8f acc[TN];
#pragma unroll
  for (int t = 0; t < TN; ++t) acc[t] = (v8f){};

  for (int k0 = 0, kt = 0; k0 < K; k0 += 32, ++kt) {
    __builtin_prefetch(arow + k0 + 32, 0, 0);
    U16 a;
    a.h[0] = *(const v8h*)(arow + k0);        // K = k0 + hf*8 + [0,8)
    a.h[1] = *(const v8h*)(arow + k0 + 16);   // K = k0 + 16 + hf*8 + [0,8)
#pragma unroll
    for (int t = 0; t < TN; ++t) {
      const _Float16* bpp = Bp + ((size_t)kt * O + t * 16 + r) * 32 + hf * 8;
      U16 b;
      b.h[0] = *(const v8h*)bpp;
      b.h[1] = *(const v8h*)(bpp + 16);
      acc[t] = __builtin_amdgcn_wmma_f32_16x16x32_f16(false, a.v, false, b.v,
                                                      (short)0, acc[t], false, false);
    }
  }
#pragma unroll
  for (int t = 0; t < TN; ++t) {
#pragma unroll
    for (int j = 0; j < 8; ++j) {
      int row = wave * 16 + hf * 8 + j;  // D: vgpr j -> M=j (lanes 0-15) / 8+j (16-31)
      C[(size_t)row * O + t * 16 + r] = acc[t][j];
    }
  }
}

// ---------------------------------------------------------------------------
// Graph-side helpers.
// ---------------------------------------------------------------------------
// mlp1 pre-BN: connections[N,4] @ w[4,16]   (bias cancels under train-mode BN)
__global__ void k_mlp1(const float* __restrict__ conn, const float* __restrict__ w,
                       float* __restrict__ out) {
  size_t idx = blockIdx.x * (size_t)blockDim.x + threadIdx.x;
  if (idx >= (size_t)NCONN * 16) return;
  int o = (int)(idx & 15);
  int n = (int)(idx >> 4);
  const float* c = conn + (size_t)n * 4;
  float s = c[0] * w[o] + c[1] * w[16 + o] + c[2] * w[32 + o] + c[3] * w[48 + o];
  out[idx] = s;
}

// Column mean / inv-std over N rows (train-mode BN1d stats).
__global__ void k_col_stats(const float* __restrict__ in, float* __restrict__ stats,
                            int N, int C) {
  int c = blockIdx.x;
  float s = 0.f, s2 = 0.f;
  for (int i = threadIdx.x; i < N; i += blockDim.x) {
    float v = in[(size_t)i * C + c];
    s += v; s2 += v * v;
  }
  __shared__ float shs[256];
  __shared__ float shq[256];
  int tid = threadIdx.x;
  shs[tid] = s; shq[tid] = s2;
  __syncthreads();
  for (int off = 128; off > 0; off >>= 1) {
    if (tid < off) { shs[tid] += shs[tid + off]; shq[tid] += shq[tid + off]; }
    __syncthreads();
  }
  if (tid == 0) {
    float m = shs[0] / (float)N;
    float v = shq[0] / (float)N - m * m;
    stats[2 * c]     = m;
    stats[2 * c + 1] = rsqrtf(v + EPSV);
  }
}

__global__ void k_bn1d_relu(float* __restrict__ x, const float* __restrict__ stats,
                            const float* __restrict__ g, const float* __restrict__ b,
                            int N, int C) {
  size_t idx = blockIdx.x * (size_t)blockDim.x + threadIdx.x;
  if (idx >= (size_t)N * C) return;
  int c = (int)(idx % C);
  float y = (x[idx] - stats[2 * c]) * stats[2 * c + 1] * g[c] + b[c];
  x[idx] = fmaxf(y, 0.f);
}

// cat[:, 0:C] = x ; cat[:, C:3C] = 0  (scatter-add targets)
__global__ void k_cat_init(const float* __restrict__ x, float* __restrict__ cat, int C) {
  size_t idx = blockIdx.x * (size_t)blockDim.x + threadIdx.x;
  size_t total = (size_t)NCONN * 3 * C;
  if (idx >= total) return;
  int j = (int)(idx % (3 * C));
  int n = (int)(idx / (3 * C));
  cat[(size_t)n * (4 * C) + j] = (j < C) ? x[(size_t)n * C + j] : 0.f;
}

// cat[edges[e,0], colOff + j] += x[edges[e,1], j]
__global__ void k_scatter(const int* __restrict__ edges, const float* __restrict__ x,
                          float* __restrict__ cat, int C, int catW, int colOff) {
  size_t idx = blockIdx.x * (size_t)blockDim.x + threadIdx.x;
  if (idx >= (size_t)NEDGE * C) return;
  int j = (int)(idx % C);
  int e = (int)(idx / C);
  int dst = edges[2 * e];
  int src = edges[2 * e + 1];
  atomicAdd(&cat[(size_t)dst * catW + colOff + j], x[(size_t)src * C + j]);
}

// Sample 8 points along each connection segment from feat[C,H,W], mean -> cat cols.
// Reference scales BOTH axes by W (bug preserved); round = nearest-even = rintf.
__global__ void k_imgfeat(const float* __restrict__ feat, const float* __restrict__ conn,
                          float* __restrict__ cat, int C, int H, int W,
                          int catW, int colOff) {
  size_t idx = blockIdx.x * (size_t)blockDim.x + threadIdx.x;
  if (idx >= (size_t)NCONN * C) return;
  int c = (int)(idx % C);
  int n = (int)(idx / C);
  float y1 = conn[(size_t)n * 4 + 0] * (float)W;
  float x1 = conn[(size_t)n * 4 + 1] * (float)W;
  float y2 = conn[(size_t)n * 4 + 2] * (float)W;
  float x2 = conn[(size_t)n * 4 + 3] * (float)W;
  const float* fc = feat + (size_t)c * H * W;
  float s = 0.f;
#pragma unroll
  for (int p = 0; p < 8; ++p) {
    float a = (float)p * (1.0f / 7.0f);
    int xs = (int)rintf(x1 + (x2 - x1) * a);
    int ys = (int)rintf(y1 + (y2 - y1) * a);
    xs = xs < 0 ? 0 : (xs > W - 1 ? W - 1 : xs);
    ys = ys < 0 ? 0 : (ys > H - 1 ? H - 1 : ys);
    s += fc[(size_t)ys * W + xs];
  }
  cat[(size_t)n * catW + colOff + c] = s * 0.125f;
}

__global__ void k_bias_relu(float* __restrict__ x, const float* __restrict__ b, int N, int O) {
  size_t idx = blockIdx.x * (size_t)blockDim.x + threadIdx.x;
  if (idx >= (size_t)N * O) return;
  int o = (int)(idx % O);
  x[idx] = fmaxf(x[idx] + b[o], 0.f);
}

// pred2: sigmoid(h[N,64] @ w[64,1] + b)
__global__ void k_pred2(const float* __restrict__ h, const float* __restrict__ w,
                        const float* __restrict__ b, float* __restrict__ out) {
  int n = blockIdx.x * blockDim.x + threadIdx.x;
  if (n >= NCONN) return;
  const float* hr = h + (size_t)n * 64;
  float s = b[0];
#pragma unroll 8
  for (int k = 0; k < 64; ++k) s += hr[k] * w[k];
  out[n] = 1.0f / (1.0f + expf(-s));
}

// ---------------------------------------------------------------------------
// Host orchestration.
// ---------------------------------------------------------------------------
extern "C" void kernel_launch(void* const* d_in, const int* in_sizes, int n_in,
                              void* d_out, int out_size, void* d_ws, size_t ws_size,
                              hipStream_t stream) {
  (void)in_sizes; (void)n_in; (void)out_size; (void)ws_size;

  // ---- input unpack: top-level dict insertion order; params = jax-pytree order
  // (dict keys sorted alphabetically at every level, lists in order).
  int ii = 0;
  const float* image        = (const float*)d_in[ii++];
  const float* connections  = (const float*)d_in[ii++];
  const int*   left_edges   = (const int*)d_in[ii++];
  const int*   right_edges  = (const int*)d_in[ii++];

  struct MlpP { const float *bn_b, *bn_g, *lin_b, *lin_w; } mlp[4];
  for (int m = 0; m < 4; ++m) {              // mlp{1..4}: bn{b,g}, lin{b,w}
    mlp[m].bn_b  = (const float*)d_in[ii++];
    mlp[m].bn_g  = (const float*)d_in[ii++];
    mlp[m].lin_b = (const float*)d_in[ii++];
    mlp[m].lin_w = (const float*)d_in[ii++];
  }
  const float* pred1_b = (const float*)d_in[ii++];
  const float* pred1_w = (const float*)d_in[ii++];
  const float* pred2_b = (const float*)d_in[ii++];
  const float* pred2_w = (const float*)d_in[ii++];

  struct BlkP {
    const float *bn1_b, *bn1_g, *bn2_b, *bn2_g, *conv1, *conv2;
    const float *ds_bn_b, *ds_bn_g, *ds_w;
    bool has_ds;
  } blk[16];
  int nblocks[4] = {3, 4, 6, 3};
  int nb = 0;
  for (int s = 0; s < 4; ++s)
    for (int b = 0; b < nblocks[s]; ++b) {
      BlkP& p = blk[nb++];
      p.bn1_b = (const float*)d_in[ii++]; p.bn1_g = (const float*)d_in[ii++];
      p.bn2_b = (const float*)d_in[ii++]; p.bn2_g = (const float*)d_in[ii++];
      p.conv1 = (const float*)d_in[ii++]; p.conv2 = (const float*)d_in[ii++];
      p.has_ds = (b == 0 && s > 0);
      if (p.has_ds) {
        p.ds_bn_b = (const float*)d_in[ii++];
        p.ds_bn_g = (const float*)d_in[ii++];
        p.ds_w    = (const float*)d_in[ii++];
      } else { p.ds_bn_b = p.ds_bn_g = p.ds_w = nullptr; }
    }
  const float* stem_bn_b = (const float*)d_in[ii++];
  const float* stem_bn_g = (const float*)d_in[ii++];
  const float* stem_w    = (const float*)d_in[ii++];

  // ---- workspace bump allocator (floats)
  float* wsf = (float*)d_ws;
  size_t cursor = 0;
  auto alloc = [&](size_t nflt) { float* p = wsf + cursor; cursor += (nflt + 63) & ~(size_t)63; return p; };

  float* stats    = alloc(512);
  float* WhF      = alloc(16384);               // 32768 halfs: max K*O (packed B)
  float* stemBuf  = alloc(16 * 256 * 256);
  float* hA       = alloc(262144);
  float* hB       = alloc(262144);
  float* T1       = alloc(262144);
  float* T2       = alloc(262144);
  float* R        = alloc(262144);
  float* feats[4];
  feats[0] = alloc(16 * 128 * 128);
  feats[1] = alloc(32 * 64 * 64);
  feats[2] = alloc(64 * 32 * 32);
  feats[3] = alloc(128 * 16 * 16);
  float* bufX     = alloc((size_t)NCONN * 128);
  float* bufCat   = alloc((size_t)NCONN * 512);
  _Float16* Bpack = (_Float16*)WhF;
  _Float16* bufCatH = (_Float16*)alloc((size_t)NCONN * 256);  // 50000*512 halfs

  // ---- CNN helpers
  auto conv = [&](const float* in, const float* w, float* out, int Cin, int Hin, int Win,
                  int Cout, int ks, int stride, int pad, int& Hout, int& Wout) {
    Hout = (Hin + 2 * pad - ks) / stride + 1;
    Wout = (Win + 2 * pad - ks) / stride + 1;
    long long total = (long long)Cout * Hout * Wout;
    k_conv<<<cdiv(total, 256), 256, 0, stream>>>(in, w, out, Cin, Hin, Win, Cout, Hout, Wout,
                                                 ks, stride, pad);
  };
  auto bn2d = [&](const float* x, const float* g, const float* b, const float* res,
                  float* out, int C, int HW, int relu) {
    k_chan_stats<<<C, 256, 0, stream>>>(x, stats, HW);
    k_bn_act<<<cdiv((long long)C * HW, 256), 256, 0, stream>>>(x, stats, g, b, res, out, C, HW, relu);
  };

  // ---- stem + pool
  int Ho, Wo;
  conv(image, stem_w, stemBuf, 4, 512, 512, 16, 7, 2, 3, Ho, Wo);              // 16x256x256
  bn2d(stemBuf, stem_bn_g, stem_bn_b, nullptr, stemBuf, 16, Ho * Wo, 1);
  k_maxpool<<<cdiv(16LL * 128 * 128, 256), 256, 0, stream>>>(stemBuf, hA, 16, 256, 256, 128, 128);

  // ---- residual stages
  int pls[4] = {16, 32, 64, 128}, strides[4] = {1, 2, 2, 2};
  float* hcur = hA;
  float* hnxt = hB;
  int H = 128, Cin = 16;
  int featH[4];
  nb = 0;
  for (int s = 0; s < 4; ++s) {
    for (int b = 0; b < nblocks[s]; ++b) {
      const BlkP& p = blk[nb++];
      int stride = (b == 0) ? strides[s] : 1;
      int pl = pls[s];
      int H1, W1, H2, W2;
      conv(hcur, p.conv1, T1, Cin, H, H, pl, 3, stride, 1, H1, W1);
      bn2d(T1, p.bn1_g, p.bn1_b, nullptr, T1, pl, H1 * W1, 1);
      conv(T1, p.conv2, T2, pl, H1, W1, pl, 3, 1, 1, H2, W2);
      const float* res = hcur;
      if (p.has_ds) {
        int Hd, Wd;
        conv(hcur, p.ds_w, R, Cin, H, H, pl, 1, stride, 0, Hd, Wd);
        bn2d(R, p.ds_bn_g, p.ds_bn_b, nullptr, R, pl, Hd * Wd, 0);
        res = R;
      }
      bn2d(T2, p.bn2_g, p.bn2_b, res, hnxt, pl, H2 * W2, 1);
      float* t = hcur; hcur = hnxt; hnxt = t;
      H = H1; Cin = pl;
    }
    k_copy<<<cdiv((long long)Cin * H * H, 256), 256, 0, stream>>>(hcur, feats[s], (size_t)Cin * H * H);
    featH[s] = H;
  }

  // ---- graph pipeline helpers
  auto bn1d_relu = [&](float* x, int C, const float* g, const float* b) {
    k_col_stats<<<C, 256, 0, stream>>>(x, stats, NCONN, C);
    k_bn1d_relu<<<cdiv((long long)NCONN * C, 256), 256, 0, stream>>>(x, stats, g, b, NCONN, C);
  };
  auto aggregate = [&](int C, const float* feat, int FH) {
    int catW = 4 * C;
    k_cat_init<<<cdiv((long long)NCONN * 3 * C, 256), 256, 0, stream>>>(bufX, bufCat, C);
    k_scatter<<<cdiv((long long)NEDGE * C, 256), 256, 0, stream>>>(left_edges,  bufX, bufCat, C, catW, C);
    k_scatter<<<cdiv((long long)NEDGE * C, 256), 256, 0, stream>>>(right_edges, bufX, bufCat, C, catW, 2 * C);
    k_imgfeat<<<cdiv((long long)NCONN * C, 256), 256, 0, stream>>>(feat, connections, bufCat, C, FH, FH, catW, 3 * C);
    k_f32_to_f16v4<<<cdiv((long long)NCONN * catW / 4, 256), 256, 0, stream>>>(
        bufCat, bufCatH, (size_t)NCONN * catW / 4);
  };
  auto wmma_gemm = [&](const float* wf, int K, int O, float* outBuf) {
    k_pack_b<<<cdiv((long long)K * O, 256), 256, 0, stream>>>(wf, Bpack, K, O);
    int blocks = cdiv((long long)(NCONN / 16) * 32, 256);
    switch (O >> 4) {
      case 2: k_wmma_strip<2><<<blocks, 256, 0, stream>>>(bufCatH, Bpack, outBuf, NCONN, K, O); break;
      case 4: k_wmma_strip<4><<<blocks, 256, 0, stream>>>(bufCatH, Bpack, outBuf, NCONN, K, O); break;
      default: k_wmma_strip<8><<<blocks, 256, 0, stream>>>(bufCatH, Bpack, outBuf, NCONN, K, O); break;
    }
  };
  auto wmma_mlp = [&](const float* wf, int K, int O, const float* g, const float* b) {
    wmma_gemm(wf, K, O, bufX);
    bn1d_relu(bufX, O, g, b);
  };

  // mlp1 (K=4 -> scalar), then 4 aggregate/MLP rounds (bias cancels under BN)
  k_mlp1<<<cdiv((long long)NCONN * 16, 256), 256, 0, stream>>>(connections, mlp[0].lin_w, bufX);
  bn1d_relu(bufX, 16, mlp[0].bn_g, mlp[0].bn_b);

  aggregate(16, feats[0], featH[0]);
  wmma_mlp(mlp[1].lin_w, 64, 32, mlp[1].bn_g, mlp[1].bn_b);

  aggregate(32, feats[1], featH[1]);
  wmma_mlp(mlp[2].lin_w, 128, 64, mlp[2].bn_g, mlp[2].bn_b);

  aggregate(64, feats[2], featH[2]);
  wmma_mlp(mlp[3].lin_w, 256, 128, mlp[3].bn_g, mlp[3].bn_b);

  aggregate(128, feats[3], featH[3]);

  // ---- prediction head: WMMA 512->64, bias+ReLU, scalar 64->1 + sigmoid
  wmma_gemm(pred1_w, 512, 64, bufX);
  k_bias_relu<<<cdiv((long long)NCONN * 64, 256), 256, 0, stream>>>(bufX, pred1_b, NCONN, 64);
  k_pred2<<<cdiv((long long)NCONN, 256), 256, 0, stream>>>(bufX, pred2_w, pred2_b, (float*)d_out);
}
